// W_LSTM_cell_29850022707330
// MI455X (gfx1250) — matrix operand order repaired and checked by reading, not compiled, not run on hardware
//
#include <hip/hip_runtime.h>

// ---------------------------------------------------------------------------
// LSTM cell, B=8192, IN=H=1024.
// bf16 WMMA GEMM (f32 accumulate) over fused K=2048 ( [x|h] @ [Wx|Wh]^T ),
// all 4 gates per workgroup, fused sigmoid/tanh/c/h epilogue.
// Staging: gfx1250 GLOBAL_LOAD_ASYNC_TO_LDS_B128 (ASYNCcnt), double-buffered
// LDS, ONE barrier per K-step.  Per wave: 2 M-blocks x (4 gates x 2 N-tiles)
// = 16 WMMAs per K-step, so each B fragment is reused twice (halves LDS
// read traffic per FLOP).  kt-loop at unroll=1 for stable acc reg homes.
// ---------------------------------------------------------------------------

typedef __bf16 bf16;
typedef __attribute__((ext_vector_type(16))) __bf16 v16bf;
typedef __attribute__((ext_vector_type(4)))  __bf16 v4bf;
typedef __attribute__((ext_vector_type(8)))  float  v8f;
typedef __attribute__((ext_vector_type(4)))  float  vf4;

struct F4x2 { vf4 a, b; };   // 32 bytes == v16bf

#define BATCH 8192
#define HID   1024
#define KTOT  2048            // IN + H
#define KSTEP 32
#define KITER (KTOT / KSTEP)  // 64
#define MT 256                // M-tile per workgroup (2 x 128 per wave set)
#define NT 32                 // N-tile per gate
#define LDS_STRIDE 80u        // padded 32-elem bf16 row (64B) -> conflict-free
// LDS map: A0 @0 (20480 B), A1 @20480, B0 @40960 (10240 B), B1 @51200
#define B_BASE 40960u
#define A_XOR  20480u
#define B_XOR  10240u

__device__ __forceinline__ float fast_sigmoid(float x) {
    return 1.0f / (1.0f + __expf(-x));
}
__device__ __forceinline__ float fast_tanh(float x) {
    return 1.0f - 2.0f / (__expf(2.0f * x) + 1.0f);
}

// Pack A = concat(x, h) along K, fp32 -> bf16.  A is [8192][2048].
__global__ void pack_A(const float* __restrict__ x, const float* __restrict__ h,
                       bf16* __restrict__ A) {
    long t = (long)blockIdx.x * blockDim.x + threadIdx.x;
    long base = t * 4;
    int row = (int)(base >> 11);
    int col = (int)(base & 2047);
    const float* src = (col < HID) ? (x + (long)row * HID + col)
                                   : (h + (long)row * HID + (col - HID));
    vf4 v = *(const vf4*)src;
    v4bf o = { (bf16)v.x, (bf16)v.y, (bf16)v.z, (bf16)v.w };
    *(v4bf*)(A + base) = o;
}

// Pack W = gate-major [4][1024][2048] bf16, gates i,f,g,o; K = [Wx | Wh].
__global__ void pack_W(const float* __restrict__ wix, const float* __restrict__ wih,
                       const float* __restrict__ wfx, const float* __restrict__ wfh,
                       const float* __restrict__ wgx, const float* __restrict__ wgh,
                       const float* __restrict__ wox, const float* __restrict__ woh,
                       bf16* __restrict__ W) {
    long t = (long)blockIdx.x * blockDim.x + threadIdx.x;
    long base = t * 4;
    int k = (int)(base & 2047);
    int rowg = (int)(base >> 11);
    int gate = rowg >> 10;
    int n = rowg & 1023;
    const float* sx;
    const float* sh;
    if      (gate == 0) { sx = wix; sh = wih; }
    else if (gate == 1) { sx = wfx; sh = wfh; }
    else if (gate == 2) { sx = wgx; sh = wgh; }
    else                { sx = wox; sh = woh; }
    const float* src = (k < HID) ? (sx + (long)n * HID + k)
                                 : (sh + (long)n * HID + (k - HID));
    vf4 v = *(const vf4*)src;
    v4bf o = { (bf16)v.x, (bf16)v.y, (bf16)v.z, (bf16)v.w };
    *(v4bf*)(W + base) = o;
}

// Fused GEMM + LSTM epilogue.
// grid = (32 M-tiles, 32 N-tiles), block = 256 (8 waves).
__launch_bounds__(256)
__global__ void lstm_gemm(const bf16* __restrict__ A, const bf16* __restrict__ W,
                          const float* __restrict__ bi, const float* __restrict__ bfv_,
                          const float* __restrict__ bg, const float* __restrict__ bo,
                          const float* __restrict__ cprev, float* __restrict__ out) {
    __shared__ __align__(16) unsigned char lds[61440];

    const int tid  = threadIdx.x;
    const int lane = tid & 31;
    const int wv   = tid >> 5;
    const int mBase = blockIdx.x * MT;
    const int nBase = blockIdx.y * NT;
    const int hi  = lane >> 4;
    const int l15 = lane & 15;

    // ---- per-thread async staging addresses (4 A chunks + 2 B chunks) ----
    const unsigned ldsBase = (unsigned)(uintptr_t)(&lds[0]);
    unsigned long long gA[4], gB[2];
    unsigned lA[4], lB[2];
#pragma unroll
    for (int i = 0; i < 4; ++i) {               // A: 1024 x 16B chunks
        const int c = tid + i * 256;
        const int row = c >> 2, kc = c & 3;
        gA[i] = (unsigned long long)(uintptr_t)
            (A + (long)(mBase + row) * KTOT + kc * 8);
        lA[i] = ldsBase + (unsigned)row * LDS_STRIDE + kc * 16;
    }
#pragma unroll
    for (int i = 0; i < 2; ++i) {               // B: 512 x 16B chunks
        const int c = tid + i * 256;
        const int row = c >> 2, kc = c & 3;
        gB[i] = (unsigned long long)(uintptr_t)
            (W + (long)((row >> 5) * HID + nBase + (row & 31)) * KTOT + kc * 8);
        lB[i] = ldsBase + B_BASE + (unsigned)row * LDS_STRIDE + kc * 16;
    }

    auto issueStage = [&]() {
#pragma unroll
        for (int i = 0; i < 4; ++i) {
            asm volatile("global_load_async_to_lds_b128 %0, %1, off"
                         :: "v"(lA[i]), "v"(gA[i]) : "memory");
        }
#pragma unroll
        for (int i = 0; i < 2; ++i) {
            asm volatile("global_load_async_to_lds_b128 %0, %1, off"
                         :: "v"(lB[i]), "v"(gB[i]) : "memory");
        }
#pragma unroll
        for (int i = 0; i < 4; ++i) { gA[i] += KSTEP * 2; lA[i] ^= A_XOR; }
#pragma unroll
        for (int i = 0; i < 2; ++i) { gB[i] += KSTEP * 2; lB[i] ^= B_XOR; }
    };

    v8f acc[16];   // [mt*8 + gate*2 + nt]
#pragma unroll
    for (int j = 0; j < 16; ++j) acc[j] = (v8f)0.0f;

    issueStage();                    // tile 0 -> buf0; pointers now at buf1

    // Compute-side LDS offsets (into `lds`), toggled each iteration.
    const unsigned aRowB = (unsigned)(wv * 16 + l15) * LDS_STRIDE + hi * 16u;
    const unsigned bRowB = B_BASE + (unsigned)l15 * LDS_STRIDE + hi * 16u;
    unsigned aOffVar = 0u;           // ^= A_XOR per iteration
    unsigned bOffVar = 0u;           // ^= B_XOR per iteration

#pragma unroll 1
    for (int kt = 0; kt < KITER; ++kt) {
        // My tile-kt copies are done; after the barrier everyone's are,
        // and everyone has finished reading the other buffer.
        asm volatile("s_wait_asynccnt 0x0" ::: "memory");
        __syncthreads();
        if (kt + 1 < KITER) issueStage();   // tile kt+1 -> other buffer

        // Two A fragments (rows wv*16.. and 128+wv*16..).
        // 16-bit 16x32 layout: lanes 0-15 K[0..7]+K[16..23],
        // lanes 16-31 K[8..15]+K[24..31].
        F4x2 a0, a1;
        a0.a = *(const vf4*)(lds + aOffVar + aRowB);
        a0.b = *(const vf4*)(lds + aOffVar + aRowB + 32);
        a1.a = *(const vf4*)(lds + aOffVar + aRowB + 10240);
        a1.b = *(const vf4*)(lds + aOffVar + aRowB + 10240 + 32);
        v16bf av0 = __builtin_bit_cast(v16bf, a0);
        v16bf av1 = __builtin_bit_cast(v16bf, a1);

        // 8 B fragments (32x16: lane=column; lanes 0-15 K[0..7]+K[16..23],
        // lanes 16-31 K[8..15]+K[24..31], matching A).
        v16bf bfrag[8];
#pragma unroll
        for (int g = 0; g < 4; ++g) {
#pragma unroll
            for (int nt = 0; nt < 2; ++nt) {
                const unsigned bo2 = bOffVar + bRowB
                                   + (unsigned)(g * 32 + nt * 16) * LDS_STRIDE;
                F4x2 t;
                t.a = *(const vf4*)(lds + bo2);
                t.b = *(const vf4*)(lds + bo2 + 32);
                bfrag[g * 2 + nt] = __builtin_bit_cast(v16bf, t);
            }
        }
        // Fence: every WMMA depends on av0/av1 as redefined HERE, below all
        // ds_loads (memory ops cannot sink past the clobber).  So all loads
        // issue first, all 8 B fragments stay live, dscnt waits pipeline.
        asm volatile("" : "+v"(av0), "+v"(av1) :: "memory");

#pragma unroll
        for (int j = 0; j < 8; ++j) {
            acc[j] = __builtin_amdgcn_wmma_f32_16x16x32_bf16(
                false, av0, false, bfrag[j], (short)0, acc[j], false, false);
            acc[8 + j] = __builtin_amdgcn_wmma_f32_16x16x32_bf16(
                false, av1, false, bfrag[j], (short)0, acc[8 + j], false, false);
        }

        aOffVar ^= A_XOR;
        bOffVar ^= B_XOR;
    }

    // ---- fused LSTM epilogue -------------------------------------------
    const float* bptr[4] = { bi, bfv_, bg, bo };
    float bias[4][2];
#pragma unroll
    for (int g = 0; g < 4; ++g)
#pragma unroll
        for (int nt = 0; nt < 2; ++nt)
            bias[g][nt] = bptr[g][nBase + nt * 16 + l15];

#pragma unroll
    for (int mt = 0; mt < 2; ++mt) {
#pragma unroll
        for (int nt = 0; nt < 2; ++nt) {
            const int n = nBase + nt * 16 + l15;
#pragma unroll
            for (int r = 0; r < 8; ++r) {
                // C/D layout: VGPR r, lanes 0-15 -> M=r, lanes 16-31 -> M=r+8
                const int m = mBase + mt * 128 + wv * 16 + r + hi * 8;
                const float zi = acc[mt * 8 + 0 * 2 + nt][r] + bias[0][nt];
                const float zf = acc[mt * 8 + 1 * 2 + nt][r] + bias[1][nt];
                const float zg = acc[mt * 8 + 2 * 2 + nt][r] + bias[2][nt];
                const float zo = acc[mt * 8 + 3 * 2 + nt][r] + bias[3][nt];
                const float ig = fast_sigmoid(zi);
                const float fg = fast_sigmoid(zf);
                const float gg = fast_tanh(zg);
                const float og = fast_sigmoid(zo);
                const long idx = (long)m * HID + n;
                const float cn = fg * cprev[idx] + ig * gg;
                const float hn = og * fast_tanh(cn);
                out[idx] = hn;                            // h_next
                out[(long)BATCH * HID + idx] = cn;        // c_next
            }
        }
    }
}

extern "C" void kernel_launch(void* const* d_in, const int* in_sizes, int n_in,
                              void* d_out, int out_size, void* d_ws, size_t ws_size,
                              hipStream_t stream) {
    (void)in_sizes; (void)n_in; (void)out_size; (void)ws_size;
    const float* x      = (const float*)d_in[0];
    const float* h      = (const float*)d_in[1];
    const float* c_prev = (const float*)d_in[2];
    const float* W_fx   = (const float*)d_in[3];
    const float* b_f    = (const float*)d_in[4];
    const float* W_fh   = (const float*)d_in[5];
    const float* W_ix   = (const float*)d_in[6];
    const float* b_i    = (const float*)d_in[7];
    const float* W_ih   = (const float*)d_in[8];
    const float* W_gx   = (const float*)d_in[9];
    const float* b_g    = (const float*)d_in[10];
    const float* W_gh   = (const float*)d_in[11];
    const float* W_ox   = (const float*)d_in[12];
    const float* b_o    = (const float*)d_in[13];
    const float* W_oh   = (const float*)d_in[14];

    bf16* Abf = (bf16*)d_ws;                                     // 32 MB
    bf16* Wbf = (bf16*)((char*)d_ws + (size_t)BATCH * KTOT * 2); // 16 MB

    {
        const int total = BATCH * KTOT / 4;
        pack_A<<<total / 256, 256, 0, stream>>>(x, h, Abf);
    }
    {
        const int total = 4 * HID * KTOT / 4;
        pack_W<<<total / 256, 256, 0, stream>>>(W_ix, W_ih, W_fx, W_fh,
                                                W_gx, W_gh, W_ox, W_oh, Wbf);
    }
    dim3 grid(BATCH / MT, HID / NT);
    lstm_gemm<<<grid, 256, 0, stream>>>(Abf, Wbf, b_i, b_f, b_g, b_o,
                                        c_prev, (float*)d_out);
}
